// DecoderBlock_1511828488676
// MI455X (gfx1250) — matrix operand order repaired and checked by reading, not compiled, not run on hardware
//
#include <hip/hip_runtime.h>

// ---------------- types / helpers ----------------
typedef unsigned int   u32;
typedef unsigned short u16;
typedef u32   u32x4   __attribute__((ext_vector_type(4)));
typedef float f32x4   __attribute__((ext_vector_type(4)));
typedef float f32x8   __attribute__((ext_vector_type(8)));
typedef __bf16 bf16x16 __attribute__((ext_vector_type(16)));

union FragU { u32x4 u[2]; bf16x16 v; };

__device__ __forceinline__ u16 f2bf(float f) {           // RNE fp32 -> bf16
  u32 u = __float_as_uint(f);
  u += 0x7FFFu + ((u >> 16) & 1u);
  return (u16)(u >> 16);
}
__device__ __forceinline__ f32x8 zero8() {
  f32x8 z = {0.f,0.f,0.f,0.f,0.f,0.f,0.f,0.f};
  return z;
}
__device__ __forceinline__ f32x8 wmma_bf16(bf16x16 a, bf16x16 b, f32x8 c) {
  return __builtin_amdgcn_wmma_f32_16x16x32_bf16(false, a, false, b, (short)0, c, false, false);
}

// ---- CDNA5 async global->LDS copy (ASYNCcnt path), with sync fallback ----
#if defined(__has_builtin)
#if __has_builtin(__builtin_amdgcn_global_load_async_to_lds_b128) && \
    __has_builtin(__builtin_amdgcn_s_wait_asynccnt)
#define USE_ASYNC_LDS 1
#endif
#endif

typedef int i32x4 __attribute__((vector_size(16)));                 // matches builtin's V4i
typedef __attribute__((address_space(1))) i32x4* as1_i32x4p;        // global
typedef __attribute__((address_space(3))) i32x4* as3_i32x4p;        // LDS (32-bit ptr)

__device__ __forceinline__ void cp16(u16* lds, const u16* g) {
#ifdef USE_ASYNC_LDS
  // AS1: same 64-bit pattern; AS3: truncation keeps addr[31:0] = LDS offset (ISA 10.2)
  __builtin_amdgcn_global_load_async_to_lds_b128(
      (as1_i32x4p)(unsigned long long)g,
      (as3_i32x4p)(unsigned int)(unsigned long long)lds, 0, 0);
#else
  *(u32x4*)lds = *(const u32x4*)g;
#endif
}
__device__ __forceinline__ void wait_async() {
#ifdef USE_ASYNC_LDS
  __builtin_amdgcn_s_wait_asynccnt(0);
#endif
}

// ---------------- weight conversion ----------------
// Fused QKV weight: dst[n][d], n = sel*1024 + h*64 + c  (sel: 0=q,1=k,2=v)
__global__ __launch_bounds__(256) void pack_qkv_w(
    const float* __restrict__ Wq, const float* __restrict__ Wk,
    const float* __restrict__ Wv, u16* __restrict__ dst) {
  u32 i = blockIdx.x * 256 + threadIdx.x;      // 3072*1024 total
  int d = i & 1023;
  int n = i >> 10;
  int sel = n >> 10;
  int nn  = n & 1023;
  int h = nn >> 6, c = nn & 63;
  const float* W = (sel == 0) ? Wq : (sel == 1) ? Wk : Wv;
  dst[i] = f2bf(W[((size_t)h * 1024 + d) * 64 + c]);
}

// src is [K][N] row-major fp32; dst is [N][K] bf16 ("BT" layout)
__global__ __launch_bounds__(256) void convT(
    const float* __restrict__ src, u16* __restrict__ dst, int N, int K) {
  size_t i = (size_t)blockIdx.x * 256 + threadIdx.x;   // N*K total
  int kk = (int)(i % K);
  int n  = (int)(i / K);
  dst[i] = f2bf(src[(size_t)kk * N + n]);
}

// ---------------- LayerNorm (row = 1024), bf16 output ----------------
__global__ __launch_bounds__(256) void ln_kernel(
    const float* __restrict__ x, const float* __restrict__ g,
    const float* __restrict__ be, u16* __restrict__ out) {
  constexpr int D = 1024;
  __shared__ float red[256];
  const int row = blockIdx.x, tid = threadIdx.x;
  const float* xr = x + (size_t)row * D;
  float lx[4];
  float s = 0.f;
#pragma unroll
  for (int i = 0; i < 4; ++i) { lx[i] = xr[tid + i * 256]; s += lx[i]; }
  red[tid] = s; __syncthreads();
  for (int off = 128; off > 0; off >>= 1) {
    if (tid < off) red[tid] += red[tid + off];
    __syncthreads();
  }
  float mu = red[0] * (1.0f / D);
  __syncthreads();
  float vs = 0.f;
#pragma unroll
  for (int i = 0; i < 4; ++i) { float d = lx[i] - mu; vs += d * d; }
  red[tid] = vs; __syncthreads();
  for (int off = 128; off > 0; off >>= 1) {
    if (tid < off) red[tid] += red[tid + off];
    __syncthreads();
  }
  float rstd = rsqrtf(red[0] * (1.0f / D) + 1e-5f);
  u16* orow = out + (size_t)row * D;
#pragma unroll
  for (int i = 0; i < 4; ++i) {
    int c = tid + i * 256;
    orow[c] = f2bf((lx[i] - mu) * rstd * g[c] + be[c]);
  }
}

// ---------------- generic WMMA GEMM (bf16 x bf16 -> fp32 acc) ----------------
// C[M,N] = A[M,K](bf16) * BT[N,K](bf16)^T  (+bias[n]) (+resid[m,n] fp32) (relu)
// Block: 256 threads = 8 waves (4x2), tile 128x128, K-step 32, double-buffered
// LDS staged with async global->LDS b128 copies.
__global__ __launch_bounds__(256) void gemm_bf16w(
    const u16* __restrict__ A, const u16* __restrict__ BT, void* __restrict__ Cout,
    const float* __restrict__ bias, const float* __restrict__ resid,
    int M, int N, int K, int relu, int out_bf16) {
  __shared__ alignas(16) u16 sA[2][128 * 32];   // [buf][row][k]
  __shared__ alignas(16) u16 sB[2][128 * 32];   // [buf][col(n)][k]
  const int tid  = threadIdx.x;
  const int wave = tid >> 5, lane = tid & 31;
  const int wm = wave >> 1, wn = wave & 1;          // 4 x 2 wave grid
  const int half = lane >> 4, l15 = lane & 15;
  const int row0 = blockIdx.x * 128;
  const int col0 = blockIdx.y * 128;

  f32x8 acc[2][4];
#pragma unroll
  for (int mt = 0; mt < 2; ++mt)
#pragma unroll
    for (int nt = 0; nt < 4; ++nt) acc[mt][nt] = zero8();

  auto stage = [&](int buf, int k0) {
#pragma unroll
    for (int it = 0; it < 2; ++it) {           // 512 16B slots per tile
      int s  = tid + it * 256;
      int r  = s >> 2, ch = (s & 3) << 3;
      cp16(&sA[buf][r * 32 + ch], &A [(size_t)(row0 + r) * K + k0 + ch]);
      cp16(&sB[buf][r * 32 + ch], &BT[(size_t)(col0 + r) * K + k0 + ch]);
    }
  };

  stage(0, 0);
  int buf = 0;
  for (int k0 = 0; k0 < K; k0 += 32) {
    wait_async();
    __syncthreads();                           // staged data for `buf` visible
    if (k0 + 32 < K) stage(buf ^ 1, k0 + 32);  // prefetch next stage behind compute

    bf16x16 af[2], bfr[4];
#pragma unroll
    for (int mt = 0; mt < 2; ++mt) {     // A frag: lane=(m,half), K = half*8+e / 16+half*8+e
      int r = wm * 32 + mt * 16 + l15;
      FragU t;
      t.u[0] = *(const u32x4*)&sA[buf][r * 32 + half * 8];
      t.u[1] = *(const u32x4*)&sA[buf][r * 32 + 16 + half * 8];
      af[mt] = t.v;
    }
#pragma unroll
    for (int nt = 0; nt < 4; ++nt) {     // B frag: lane=(n,half), K = half*8+e / 16+half*8+e
      int r = wn * 64 + nt * 16 + l15;
      FragU t;
      t.u[0] = *(const u32x4*)&sB[buf][r * 32 + half * 8];
      t.u[1] = *(const u32x4*)&sB[buf][r * 32 + 16 + half * 8];
      bfr[nt] = t.v;
    }
#pragma unroll
    for (int mt = 0; mt < 2; ++mt)
#pragma unroll
      for (int nt = 0; nt < 4; ++nt)
        acc[mt][nt] = wmma_bf16(af[mt], bfr[nt], acc[mt][nt]);

    __syncthreads();                           // all reads of `buf` done
    buf ^= 1;
  }

  // epilogue; C/D layout: lane N=l15, M = half*8 + r
#pragma unroll
  for (int mt = 0; mt < 2; ++mt) {
#pragma unroll
    for (int nt = 0; nt < 4; ++nt) {
      int gmb = row0 + wm * 32 + mt * 16 + half * 8;
      int gn  = col0 + wn * 64 + nt * 16 + l15;
      float bv = bias ? bias[gn] : 0.0f;
#pragma unroll
      for (int r = 0; r < 8; ++r) {
        size_t idx = (size_t)(gmb + r) * N + gn;
        float v = acc[mt][nt][r] + bv;
        if (resid) v += resid[idx];
        if (relu)  v = fmaxf(v, 0.0f);
        if (out_bf16) ((u16*)Cout)[idx] = f2bf(v);
        else          ((float*)Cout)[idx] = v;
      }
    }
  }
}

// ---------------- repack qkv [B*S][3072] bf16 -> q,k [BH][S][64], vT [BH][64][S] ----------------
__global__ __launch_bounds__(256) void repack_qkv(
    const u16* __restrict__ qkv, u16* __restrict__ q,
    u16* __restrict__ k, u16* __restrict__ vT) {
  u32 i = blockIdx.x * 256 + threadIdx.x;   // 4*1024*16*64 total
  int c = i & 63;
  int h = (i >> 6) & 15;
  int s = (i >> 10) & 1023;
  int b = i >> 20;
  size_t row = (size_t)(b * 1024 + s) * 3072;
  int bh = b * 16 + h;
  q [((size_t)bh * 1024 + s) * 64 + c] = qkv[row + h * 64 + c];
  k [((size_t)bh * 1024 + s) * 64 + c] = qkv[row + 1024 + h * 64 + c];
  vT[((size_t)bh * 64 + c) * 1024 + s] = qkv[row + 2048 + h * 64 + c];
}

// ---------------- causal flash attention (bf16 ctx output) ----------------
// grid (S/64, B*H), block 128 (4 waves); each wave: 16 query rows, key blocks of 32
__global__ __launch_bounds__(128) void attn_kernel(
    const u16* __restrict__ q, const u16* __restrict__ k,
    const u16* __restrict__ vT, u16* __restrict__ ctx) {
  constexpr int S = 1024;
  __shared__ alignas(16) u16 sP[4][16 * 32];
  const int wave = threadIdx.x >> 5, lane = threadIdx.x & 31;
  const int half = lane >> 4, l15 = lane & 15;
  const int bh = blockIdx.y;
  const int b = bh >> 4, h = bh & 15;
  const int s0 = (blockIdx.x * 4 + wave) * 16;
  const u16* qb = q  + (size_t)bh * S * 64;
  const u16* kb = k  + (size_t)bh * S * 64;
  const u16* vb = vT + (size_t)bh * 64 * S;
  u16* myP = sP[wave];

  bf16x16 aq[2];                           // q rows s0..s0+15, K-slices of 32
#pragma unroll
  for (int kk = 0; kk < 2; ++kk) {
    const u16* p = &qb[(size_t)(s0 + l15) * 64 + kk * 32 + half * 8];
    FragU t; t.u[0] = *(const u32x4*)p; t.u[1] = *(const u32x4*)(p + 16);
    aq[kk] = t.v;
  }

  float m_i[8], l_i[8];
#pragma unroll
  for (int r = 0; r < 8; ++r) { m_i[r] = -__builtin_inff(); l_i[r] = 0.f; }
  f32x8 accO[4];
#pragma unroll
  for (int nt = 0; nt < 4; ++nt) accO[nt] = zero8();

  for (int t0 = 0; t0 < s0 + 16; t0 += 32) {
    float pv[2][8];
    float mloc[8];
#pragma unroll
    for (int r = 0; r < 8; ++r) mloc[r] = -__builtin_inff();

#pragma unroll
    for (int tt = 0; tt < 2; ++tt) {
      f32x8 sc = zero8();
#pragma unroll
      for (int kk = 0; kk < 2; ++kk) {     // B frag from k (BT layout: rows of dk)
        const u16* bp = &kb[(size_t)(t0 + tt * 16 + l15) * 64 + kk * 32 + half * 8];
        FragU t; t.u[0] = *(const u32x4*)bp; t.u[1] = *(const u32x4*)(bp + 16);
        sc = wmma_bf16(aq[kk], t.v, sc);
      }
      int tcol = t0 + tt * 16 + l15;
#pragma unroll
      for (int r = 0; r < 8; ++r) {
        int gm = s0 + half * 8 + r;
        float sv = sc[r] * 0.125f;          // 1/sqrt(64)
        if (tcol > gm) sv = -__builtin_inff();
        pv[tt][r] = sv;
        mloc[r] = fmaxf(mloc[r], sv);
      }
    }

    // online softmax: row reductions across the 16 lanes of each half
#pragma unroll
    for (int r = 0; r < 8; ++r) {
      float m = mloc[r];
      m = fmaxf(m, __shfl_xor(m, 1, 32));
      m = fmaxf(m, __shfl_xor(m, 2, 32));
      m = fmaxf(m, __shfl_xor(m, 4, 32));
      m = fmaxf(m, __shfl_xor(m, 8, 32));
      float mnew  = fmaxf(m_i[r], m);
      float alpha = __expf(m_i[r] - mnew);
      m_i[r] = mnew;
      float rs = 0.f;
#pragma unroll
      for (int tt = 0; tt < 2; ++tt) {
        float e = __expf(pv[tt][r] - mnew);
        pv[tt][r] = e;
        rs += e;
      }
      rs += __shfl_xor(rs, 1, 32);
      rs += __shfl_xor(rs, 2, 32);
      rs += __shfl_xor(rs, 4, 32);
      rs += __shfl_xor(rs, 8, 32);
      l_i[r] = l_i[r] * alpha + rs;
#pragma unroll
      for (int nt = 0; nt < 4; ++nt) accO[nt][r] *= alpha;
    }

    // P: C-layout -> LDS row-major [16 rows][32 t] -> A-layout fragment
#pragma unroll
    for (int tt = 0; tt < 2; ++tt)
#pragma unroll
      for (int r = 0; r < 8; ++r)
        myP[(half * 8 + r) * 32 + tt * 16 + l15] = f2bf(pv[tt][r]);

    FragU tp;
    tp.u[0] = *(const u32x4*)&myP[l15 * 32 + half * 8];
    tp.u[1] = *(const u32x4*)&myP[l15 * 32 + 16 + half * 8];
    bf16x16 aP = tp.v;

#pragma unroll
    for (int nt = 0; nt < 4; ++nt) {       // B frag from vT (BT layout: rows of t)
      const u16* bp = &vb[(size_t)(nt * 16 + l15) * S + t0 + half * 8];
      FragU t; t.u[0] = *(const u32x4*)bp; t.u[1] = *(const u32x4*)(bp + 16);
      accO[nt] = wmma_bf16(aP, t.v, accO[nt]);
    }
  }

#pragma unroll
  for (int r = 0; r < 8; ++r) l_i[r] = 1.0f / l_i[r];
#pragma unroll
  for (int nt = 0; nt < 4; ++nt)
#pragma unroll
    for (int r = 0; r < 8; ++r) {
      int gm = s0 + half * 8 + r;
      int gn = h * 64 + nt * 16 + l15;
      ctx[(size_t)(b * S + gm) * 1024 + gn] = f2bf(accO[nt][r] * l_i[r]);
    }
}

// ---------------- host launcher ----------------
extern "C" void kernel_launch(void* const* d_in, const int* in_sizes, int n_in,
                              void* d_out, int out_size, void* d_ws, size_t ws_size,
                              hipStream_t stream) {
  (void)in_sizes; (void)n_in; (void)out_size; (void)ws_size;
  const float* x   = (const float*)d_in[0];
  const float* Wq  = (const float*)d_in[1];
  const float* Wk  = (const float*)d_in[2];
  const float* Wv  = (const float*)d_in[3];
  const float* Wo  = (const float*)d_in[4];
  const float* w1  = (const float*)d_in[5];
  const float* b1  = (const float*)d_in[6];
  const float* w2  = (const float*)d_in[7];
  const float* b2  = (const float*)d_in[8];
  const float* g1  = (const float*)d_in[9];
  const float* be1 = (const float*)d_in[10];
  const float* g2  = (const float*)d_in[11];
  const float* be2 = (const float*)d_in[12];
  float* out = (float*)d_out;

  // workspace layout (104 MB total; hbuf aliases the dead qkv/q region)
  char* ws = (char*)d_ws;
  constexpr size_t MB = 1ull << 20;
  u16*   nbufb = (u16*)(ws + 0);            //  8 MB  LN output bf16 (reused)
  u16*   qkv   = (u16*)(ws + 8 * MB);       // 24 MB  fused qkv (bf16)
  u16*   qb    = (u16*)(ws + 32 * MB);      //  8 MB
  u16*   kb    = (u16*)(ws + 40 * MB);      //  8 MB
  u16*   vT    = (u16*)(ws + 48 * MB);      //  8 MB
  u16*   ctxb  = (u16*)(ws + 56 * MB);      //  8 MB  attention out (bf16)
  float* x2    = (float*)(ws + 64 * MB);    // 16 MB  fp32 residual stream
  u16*   hbufb = (u16*)(ws + 8 * MB);       // 32 MB  FFN hidden bf16 (alias 8..40)
  u16*   WqkvT = (u16*)(ws + 80 * MB);      //  6 MB
  u16*   WoT   = (u16*)(ws + 86 * MB);      //  2 MB
  u16*   w1T   = (u16*)(ws + 88 * MB);      //  8 MB
  u16*   w2T   = (u16*)(ws + 96 * MB);      //  8 MB -> 104 MB

  // weights -> bf16 BT layout
  pack_qkv_w<<<(3072 * 1024) / 256, 256, 0, stream>>>(Wq, Wk, Wv, WqkvT);
  convT<<<(1024 * 1024) / 256, 256, 0, stream>>>(Wo, WoT, 1024, 1024);
  convT<<<(4096 * 1024) / 256, 256, 0, stream>>>(w1, w1T, 4096, 1024);
  convT<<<(4096 * 1024) / 256, 256, 0, stream>>>(w2, w2T, 1024, 4096);

  // n1 = LN(x) (bf16)
  ln_kernel<<<4096, 256, 0, stream>>>(x, g1, be1, nbufb);
  // fused qkv = n1 @ Wqkv   (bf16 output)
  gemm_bf16w<<<dim3(32, 24), 256, 0, stream>>>(nbufb, WqkvT, qkv,
      nullptr, nullptr, 4096, 3072, 1024, 0, 1);
  repack_qkv<<<(4 * 1024 * 16 * 64) / 256, 256, 0, stream>>>(qkv, qb, kb, vT);
  // attention -> ctx [B,S,H*64] bf16
  attn_kernel<<<dim3(16, 64), 128, 0, stream>>>(qb, kb, vT, ctxb);
  // x2 = x + ctx @ Wo   (fp32)
  gemm_bf16w<<<dim3(32, 8), 256, 0, stream>>>(ctxb, WoT, x2,
      nullptr, x, 4096, 1024, 1024, 0, 0);
  // n2 = LN(x2) (bf16)
  ln_kernel<<<4096, 256, 0, stream>>>(x2, g2, be2, nbufb);
  // h = relu(n2 @ w1 + b1)  (bf16)
  gemm_bf16w<<<dim3(32, 32), 256, 0, stream>>>(nbufb, w1T, hbufb,
      b1, nullptr, 4096, 4096, 1024, 1, 1);
  // out = x2 + h @ w2 + b2  (fp32)
  gemm_bf16w<<<dim3(32, 8), 256, 0, stream>>>(hbufb, w2T, out,
      b2, x2, 4096, 1024, 4096, 0, 0);
}
